// Attention_28389733827124
// MI455X (gfx1250) — matrix operand order repaired and checked by reading, not compiled, not run on hardware
//
#include <hip/hip_runtime.h>
#include <hip/hip_bf16.h>
#include <math.h>

#define SEQ 4096
#define HID 2048
#define NH  16
#define NKV 2
#define HD  128
#define KVW (NKV*HD)   // 256

typedef __attribute__((ext_vector_type(16))) _Float16     v16h;
typedef __attribute__((ext_vector_type(8)))  float        v8f;
typedef __attribute__((ext_vector_type(8)))  unsigned int v8u;
typedef __attribute__((ext_vector_type(4)))  int          v4i;

// Address-space-qualified int4 for the async global->LDS builtin
// (probe: param 0 is int4* in global AS, param 1 is int4* in LDS AS)
typedef __attribute__((address_space(1))) v4i gas_v4i;
typedef __attribute__((address_space(3))) v4i las_v4i;

#if defined(__has_builtin)
#if __has_builtin(__builtin_amdgcn_global_load_async_to_lds_b128)
#define HAVE_ASYNC_LDS 1
#endif
#if __has_builtin(__builtin_amdgcn_s_wait_asynccnt)
#define HAVE_WAIT_ASYNC_BUILTIN 1
#endif
#endif

__device__ __forceinline__ void wait_async_zero() {
#if defined(HAVE_ASYNC_LDS)
#if defined(HAVE_WAIT_ASYNC_BUILTIN)
  __builtin_amdgcn_s_wait_asynccnt(0);
#else
  asm volatile("s_wait_asynccnt 0x0" ::: "memory");
#endif
#endif
}

// A-matrix f16 fragment (16x32): per lane-half, K chunks [k0,k0+8) and [k0+16,k0+24),
// k0 = 8*(lane>=16). 'row' points at the lane's row (16B aligned, stride mult of 8 elems).
__device__ __forceinline__ v16h frag_ld_a(const _Float16* row, int k0) {
  uint4 lo = *(const uint4*)(row + k0);
  uint4 hi = *(const uint4*)(row + k0 + 16);
  v8u r = { lo.x, lo.y, lo.z, lo.w, hi.x, hi.y, hi.z, hi.w };
  return __builtin_bit_cast(v16h, r);
}
// B-matrix f16 fragment (32x16): per lane-half, K chunks [k0,k0+8) and [k0+16,k0+24), k0 = 8*(lane>=16).
// 'row' points at the lane's column of B stored as [n][k].
__device__ __forceinline__ v16h frag_ld_b(const _Float16* row, int k0) {
  uint4 lo = *(const uint4*)(row + k0);
  uint4 hi = *(const uint4*)(row + k0 + 16);
  v8u r = { lo.x, lo.y, lo.z, lo.w, hi.x, hi.y, hi.z, hi.w };
  return __builtin_bit_cast(v16h, r);
}

// ---------------------------------------------------------------------------
// C[M,N] = A[M,K] @ B[K,N] (+ bias). 128x128 block tile, BK=32, 256 thr / 8 waves.
// f32 global operands converted to f16 during LDS staging; v_wmma_f32_16x16x32_f16.
// Next-tile global_prefetch_b8 issued during staging to overlap with the WMMAs.
// ---------------------------------------------------------------------------
__global__ __launch_bounds__(256)
void gemm_f32_wmma(const float* __restrict__ A, const float* __restrict__ B,
                   const float* __restrict__ bias, float* __restrict__ C,
                   int M, int K, int N)
{
  __shared__ alignas(16) _Float16 As[128*40];   // [row][k], pad 32->40
  __shared__ alignas(16) _Float16 Bs[128*40];   // [n][k] (transposed), pad 32->40
  const int tid  = threadIdx.x;
  const int lane = tid & 31, wid = tid >> 5;
  const int lrow = lane & 15, lhalf = lane >> 4;
  const int wm = (wid >> 1) * 32, wn = (wid & 1) * 64;
  const int m0 = blockIdx.y * 128, n0 = blockIdx.x * 128;

  v8f acc[2][4] = {};

  for (int kk = 0; kk < K; kk += 32) {
    __syncthreads();
    const bool more = (kk + 32) < K;
    // stage A: 128x32 f32 -> f16
#pragma unroll
    for (int i = 0; i < 4; ++i) {
      int id = tid + 256*i;
      int r = id >> 3, c = (id & 7) << 2;
      const float* src = A + (size_t)(m0 + r) * K + kk + c;
      const float4 v = *(const float4*)src;
      if (more) __builtin_prefetch(src + 32, 0, 3);   // next K-chunk -> global_prefetch_b8
      _Float16* d = As + r*40 + c;
      d[0] = (_Float16)v.x; d[1] = (_Float16)v.y;
      d[2] = (_Float16)v.z; d[3] = (_Float16)v.w;
    }
    // stage B: 32x128 f32 -> f16, transposed into Bs[n][k]
#pragma unroll
    for (int i = 0; i < 4; ++i) {
      int id = tid + 256*i;
      int kr = id >> 5, c = (id & 31) << 2;
      const float* src = B + (size_t)(kk + kr) * N + n0 + c;
      const float4 v = *(const float4*)src;
      if (more) __builtin_prefetch(src + (size_t)32 * N, 0, 3);
      Bs[(c+0)*40 + kr] = (_Float16)v.x;
      Bs[(c+1)*40 + kr] = (_Float16)v.y;
      Bs[(c+2)*40 + kr] = (_Float16)v.z;
      Bs[(c+3)*40 + kr] = (_Float16)v.w;
    }
    __syncthreads();

    v16h af[2];
#pragma unroll
    for (int mi = 0; mi < 2; ++mi)
      af[mi] = frag_ld_a(As + (wm + mi*16 + lrow)*40, lhalf*8);
#pragma unroll
    for (int ni = 0; ni < 4; ++ni) {
      v16h bf = frag_ld_b(Bs + (wn + ni*16 + lrow)*40, lhalf*8);
#pragma unroll
      for (int mi = 0; mi < 2; ++mi)
        acc[mi][ni] = __builtin_amdgcn_wmma_f32_16x16x32_f16(
            false, af[mi], false, bf, (short)0, acc[mi][ni], false, false);
    }
  }

#pragma unroll
  for (int ni = 0; ni < 4; ++ni) {
    int col = n0 + wn + ni*16 + lrow;
    float b = bias ? bias[col] : 0.0f;
#pragma unroll
    for (int mi = 0; mi < 2; ++mi) {
#pragma unroll
      for (int r = 0; r < 8; ++r) {
        int row = m0 + wm + mi*16 + r + 8*lhalf;   // C layout: m = vgpr + 8*(lane>=16)
        C[(size_t)row * N + col] = acc[mi][ni][r] + b;
      }
    }
  }
}

// ---------------------------------------------------------------------------
// RoPE on Q (16 heads) and K (2 kv heads), f32 -> f16. cos/sin are [S,128].
// ---------------------------------------------------------------------------
__global__ __launch_bounds__(256)
void rope_convert(const float* __restrict__ qf, const float* __restrict__ kf,
                  const float* __restrict__ cosb, const float* __restrict__ sinb,
                  _Float16* __restrict__ qh, _Float16* __restrict__ kh)
{
  int idx = blockIdx.x * 256 + threadIdx.x;           // SEQ * 18 * 128 total
  int d  = idx & 127;
  int t  = idx >> 7;
  int hh = t % 18;
  int s  = t / 18;
  float c  = cosb[s*HD + d];
  float sn = sinb[s*HD + d];
  if (hh < NH) {
    const float* p = qf + (size_t)s*HID + hh*HD;
    float v   = p[d];
    float rot = (d < 64) ? -p[d + 64] : p[d - 64];
    qh[(size_t)s*HID + hh*HD + d] = (_Float16)(v*c + rot*sn);
  } else {
    int kvh = hh - NH;
    const float* p = kf + (size_t)s*KVW + kvh*HD;
    float v   = p[d];
    float rot = (d < 64) ? -p[d + 64] : p[d - 64];
    kh[(size_t)s*KVW + kvh*HD + d] = (_Float16)(v*c + rot*sn);
  }
}

// ---------------------------------------------------------------------------
// Flash attention: grid (S/128, NH); 256 thr / 8 waves; each wave owns 16 queries.
// Online softmax in registers; QK^T and P@V via v_wmma_f32_16x16x32_f16.
// K tile staged via async global->LDS (ASYNCcnt) when the toolchain exposes it.
// ---------------------------------------------------------------------------
__global__ __launch_bounds__(256)
void flash_attn_wmma(const _Float16* __restrict__ Qh, const _Float16* __restrict__ Kh,
                     const float* __restrict__ Vf, float* __restrict__ Of)
{
  __shared__ alignas(16) _Float16 Ks[32*136];    // [key][d], pad 128->136
  __shared__ alignas(16) _Float16 Vt[128*40];    // [d][key], pad 32->40
  __shared__ alignas(16) _Float16 Pb[8*16*40];   // per-wave P buffer [m][k], pad 32->40

  const int tid  = threadIdx.x;
  const int lane = tid & 31, wid = tid >> 5;
  const int lrow = lane & 15, lhalf = lane >> 4;
  const int h = blockIdx.y, kvh = h >> 3;        // 8 query heads per kv head
  const int qbase = blockIdx.x * 128 + wid * 16;

  // Q fragments (A layout) loaded straight from global, kept in registers
  v16h qf[4];
  const _Float16* qrow = Qh + (size_t)(qbase + lrow) * HID + h * HD;
#pragma unroll
  for (int kb = 0; kb < 4; ++kb)
    qf[kb] = frag_ld_a(qrow, kb*32 + lhalf*8);

  v8f o[8] = {};
  float mrun[8], lrun[8];
#pragma unroll
  for (int r = 0; r < 8; ++r) { mrun[r] = -1e30f; lrun[r] = 0.0f; }

  _Float16* myP = Pb + wid * (16*40);
  const int nkt = blockIdx.x * 4 + 4;            // causal: keys up to block's last query
  const float scale = 0.08838834764831845f;      // 1/sqrt(128)

  for (int kt = 0; kt < nkt; ++kt) {
    __syncthreads();
    // stage K tile (32 keys x 128 d), row-major f16: async DMA to LDS if available
#pragma unroll
    for (int i = 0; i < 2; ++i) {
      int id = tid + 256*i;
      int key = id >> 4, c8 = id & 15;
      const _Float16* src = Kh + (size_t)(kt*32 + key) * KVW + kvh*HD + c8*8;
      _Float16* dst = Ks + key*136 + c8*8;
#if defined(HAVE_ASYNC_LDS)
      __builtin_amdgcn_global_load_async_to_lds_b128(
          (gas_v4i*)src, (las_v4i*)dst, 0, 0);
#else
      *(uint4*)dst = *(const uint4*)src;
#endif
    }
    // stage V tile transposed: Vt[d][key], f32 -> f16 (needs VALU convert)
    {
      int key = tid >> 3, d0 = (tid & 7) << 4;
      const float* vs = Vf + (size_t)(kt*32 + key) * KVW + kvh*HD + d0;
#pragma unroll
      for (int j = 0; j < 16; ++j)
        Vt[(d0 + j)*40 + key] = (_Float16)vs[j];
    }
    wait_async_zero();     // wave's async K copies landed in LDS
    __syncthreads();       // publish K/V tiles workgroup-wide

    const bool active = (kt*32) <= (qbase + 15); // skip fully-masked tiles (uniform per wave)
    if (active) {
      v8f s0 = {}, s1 = {};
#pragma unroll
      for (int st = 0; st < 2; ++st) {           // two 16-key subtiles
        v8f c = {};
#pragma unroll
        for (int kb = 0; kb < 4; ++kb) {         // d = 128 in K=32 chunks
          v16h bf = frag_ld_b(Ks + (st*16 + lrow)*136, kb*32 + lhalf*8);
          c = __builtin_amdgcn_wmma_f32_16x16x32_f16(false, qf[kb], false, bf,
                                                     (short)0, c, false, false);
        }
        int keyg = kt*32 + st*16 + lrow;
#pragma unroll
        for (int r = 0; r < 8; ++r) {
          int qg = qbase + r + 8*lhalf;
          float sv = c[r] * scale;
          c[r] = (keyg > qg) ? -1e30f : sv;      // causal mask
        }
        if (st == 0) s0 = c; else s1 = c;
      }
      // row-wise online softmax; rows live across 16-lane halves at fixed vgpr idx
      float tmax[8], lsum[8], alpha[8];
#pragma unroll
      for (int r = 0; r < 8; ++r) tmax[r] = fmaxf(s0[r], s1[r]);
#pragma unroll
      for (int m = 1; m < 16; m <<= 1)
#pragma unroll
        for (int r = 0; r < 8; ++r)
          tmax[r] = fmaxf(tmax[r], __shfl_xor(tmax[r], m, 32));
#pragma unroll
      for (int r = 0; r < 8; ++r) {
        float mn = fmaxf(mrun[r], tmax[r]);
        alpha[r] = __expf(mrun[r] - mn);
        float p0 = __expf(s0[r] - mn);
        float p1 = __expf(s1[r] - mn);
        s0[r] = p0; s1[r] = p1;
        lsum[r] = p0 + p1;
        mrun[r] = mn;
      }
#pragma unroll
      for (int m = 1; m < 16; m <<= 1)
#pragma unroll
        for (int r = 0; r < 8; ++r)
          lsum[r] += __shfl_xor(lsum[r], m, 32);
#pragma unroll
      for (int r = 0; r < 8; ++r) lrun[r] = lrun[r]*alpha[r] + lsum[r];
#pragma unroll
      for (int dt = 0; dt < 8; ++dt)
#pragma unroll
        for (int r = 0; r < 8; ++r) o[dt][r] *= alpha[r];
      // write P (f16) to LDS in [m][k] layout for A-fragment re-read
#pragma unroll
      for (int r = 0; r < 8; ++r) {
        int m = r + 8*lhalf;
        myP[m*40 + lrow]      = (_Float16)s0[r];
        myP[m*40 + 16 + lrow] = (_Float16)s1[r];
      }
    }
    __syncthreads();   // publish P (and keep workgroup converged)
    if (active) {
      v16h pa = frag_ld_a(myP + lrow*40, lhalf*8);
#pragma unroll
      for (int dt = 0; dt < 8; ++dt) {
        v16h vb = frag_ld_b(Vt + (dt*16 + lrow)*40, lhalf*8);
        o[dt] = __builtin_amdgcn_wmma_f32_16x16x32_f16(false, pa, false, vb,
                                                       (short)0, o[dt], false, false);
      }
    }
  }

#pragma unroll
  for (int dt = 0; dt < 8; ++dt) {
    int col = h*HD + dt*16 + lrow;
#pragma unroll
    for (int r = 0; r < 8; ++r) {
      int row = qbase + r + 8*lhalf;
      Of[(size_t)row * HID + col] = o[dt][r] / lrun[r];
    }
  }
}

// ---------------------------------------------------------------------------
extern "C" void kernel_launch(void* const* d_in, const int* in_sizes, int n_in,
                              void* d_out, int out_size, void* d_ws, size_t ws_size,
                              hipStream_t stream)
{
  const float* x    = (const float*)d_in[0];
  const float* cosb = (const float*)d_in[1];
  const float* sinb = (const float*)d_in[2];
  const float* Wq   = (const float*)d_in[3];
  const float* bq   = (const float*)d_in[4];
  const float* Wk   = (const float*)d_in[5];
  const float* bk   = (const float*)d_in[6];
  const float* Wv   = (const float*)d_in[7];
  const float* bv   = (const float*)d_in[8];
  const float* Wo   = (const float*)d_in[9];
  float* out = (float*)d_out;

  char* ws = (char*)d_ws;
  float*    qf = (float*)(ws);                                       // 33.5 MB
  float*    kf = (float*)(ws + 33554432);                            //  4.2 MB
  float*    vf = (float*)(ws + 33554432 + 4194304);                  //  4.2 MB
  _Float16* qh = (_Float16*)(ws + 33554432 + 2*4194304);             // 16.8 MB
  _Float16* kh = (_Float16*)(ws + 33554432 + 2*4194304 + 16777216);  //  2.1 MB
  float* attn = qf;   // q f32 is dead after RoPE; reuse for attention output

  dim3 blk(256);
  gemm_f32_wmma<<<dim3(16, 32), blk, 0, stream>>>(x, Wq, bq, qf, SEQ, HID, HID);
  gemm_f32_wmma<<<dim3(2,  32), blk, 0, stream>>>(x, Wk, bk, kf, SEQ, HID, KVW);
  gemm_f32_wmma<<<dim3(2,  32), blk, 0, stream>>>(x, Wv, bv, vf, SEQ, HID, KVW);
  rope_convert<<<dim3((SEQ*18*HD)/256), blk, 0, stream>>>(qf, kf, cosb, sinb, qh, kh);
  flash_attn_wmma<<<dim3(SEQ/128, NH), blk, 0, stream>>>(qh, kh, vf, attn);
  gemm_f32_wmma<<<dim3(16, 32), blk, 0, stream>>>(attn, Wo, nullptr, out, SEQ, HID, HID);
}
